// AdaptiveSoftmax_55972013802337
// MI455X (gfx1250) — matrix operand order repaired and checked by reading, not compiled, not run on hardware
//
#include <hip/hip_runtime.h>
#include <hip/hip_bf16.h>
#include <stdint.h>

// ---------------------------------------------------------------------------
// Adaptive softmax for MI455X (gfx1250, wave32, WMMA).
// GEMMs run on v_wmma_f32_16x16x32_bf16 with f32 accumulation; all bf16
// operands (<32 MB total) are L2-resident on the 192 MB L2, so fragments are
// loaded directly from global in the ISA-documented WMMA register layouts.
// Wave tile 32(M) x 64(N): 8 WMMAs per K-step from 2 A-frags + 4 B-frags
// (1.5 b128 loads per WMMA) to keep the XDL pipes fed near peak.
// ---------------------------------------------------------------------------

typedef __attribute__((ext_vector_type(16))) __bf16 v16bf;
typedef __attribute__((ext_vector_type(8)))  float  v8f;

#define TOKENS 2048      // B*S = 2*1024
#define IN_DIM 512
#define V0 20000
#define V1 20000
#define V2 10257
#define VT 50257         // V0+V1+V2

static __device__ __forceinline__ uint16_t f32_to_bf16_bits(float f) {
  uint32_t u = __float_as_uint(f);
  uint32_t r = u + 0x7FFFu + ((u >> 16) & 1u);   // round-to-nearest-even
  return (uint16_t)(r >> 16);
}
static __device__ __forceinline__ float bf16_bits_to_f32(uint16_t h) {
  return __uint_as_float(((uint32_t)h) << 16);
}

// ---------------------------------------------------------------------------
// fp32 -> bf16 conversion
// ---------------------------------------------------------------------------
__global__ void cvt_f32_to_bf16(const float* __restrict__ src,
                                uint16_t* __restrict__ dst, int n) {
  int i = blockIdx.x * blockDim.x + threadIdx.x;
  if (i < n) dst[i] = f32_to_bf16_bits(src[i]);
}

// ---------------------------------------------------------------------------
// NT GEMM: C[M,N] = A[M,K] * B[N,K]^T   (A,B bf16, accumulate f32)
// Block: 256 threads = 8 waves, block tile 64(M) x 256(N).
// Wave grid 2(M) x 4(N); each wave owns a 32x64 tile = 2x4 WMMA 16x16 tiles.
// Epilogue: Cf != nullptr -> write f32 + bias (logits into d_out slice)
//           else          -> write bf16      (hidden h into workspace)
// M must be a multiple of 64 (2048 is); N is bounds-checked.
// ---------------------------------------------------------------------------
__global__ __launch_bounds__(256) void gemm_nt_bf16(
    const uint16_t* __restrict__ A,   // M x K, row-major
    const uint16_t* __restrict__ B,   // N x K, row-major
    int N, int K,
    float* __restrict__ Cf,           // f32 output (or nullptr)
    uint16_t* __restrict__ Cb,        // bf16 output (or nullptr)
    const float* __restrict__ bias,   // length N (used only when Cf)
    int ldc)                          // leading dim of C in elements
{
  const int lane    = threadIdx.x & 31;
  const int wave    = threadIdx.x >> 5;
  const int wm      = wave & 1;        // wave M index  (0..1)
  const int wn      = wave >> 1;       // wave N index  (0..3)
  const int blockM  = blockIdx.y * 64;
  const int blockN  = blockIdx.x * 256;
  const int halfSel = lane >> 4;       // which half-wave
  const int l15     = lane & 15;

  // ISA §7.12.2 16-bit layouts:
  //  A 16x32: lanes 0-15 hold K {0..7, 16..23}; lanes 16-31 hold K {8..15, 24..31}
  //  B 32x16: lanes 0-15 hold K {0..7, 16..23}; lanes 16-31 hold K {8..15, 24..31}
  const int aK0 = halfSel * 8;    // + second 16B chunk at +16 elements
  const int bK0 = halfSel * 8;    // + second 16B chunk at +16 elements

  // per-lane base pointers for the 2 A-fragment rows and 4 B-fragment rows
  const uint16_t* pa[2];
#pragma unroll
  for (int m = 0; m < 2; ++m) {
    const int aRow = blockM + wm * 32 + m * 16 + l15;     // always in range
    pa[m] = A + (size_t)aRow * K + aK0;
  }
  const uint16_t* pb[4];
#pragma unroll
  for (int n = 0; n < 4; ++n) {
    int bRow = blockN + wn * 64 + n * 16 + l15;
    if (bRow > N - 1) bRow = N - 1;                       // ragged vocab edge
    pb[n] = B + (size_t)bRow * K + bK0;
  }

  v8f acc[2][4];
#pragma unroll
  for (int m = 0; m < 2; ++m)
#pragma unroll
    for (int n = 0; n < 4; ++n) acc[m][n] = (v8f){};

  for (int kk = 0; kk < K; kk += 32) {
    union { v16bf v; uint32_t u[8]; } aF[2], bF[4];

#pragma unroll
    for (int m = 0; m < 2; ++m) {
      ((uint4*)aF[m].u)[0] = *(const uint4*)(pa[m] + kk);
      ((uint4*)aF[m].u)[1] = *(const uint4*)(pa[m] + kk + 16);
    }
#pragma unroll
    for (int n = 0; n < 4; ++n) {
      ((uint4*)bF[n].u)[0] = *(const uint4*)(pb[n] + kk);
      ((uint4*)bF[n].u)[1] = *(const uint4*)(pb[n] + kk + 16);
    }

    // speculative prefetch of next K-slice (global_prefetch_b8; OOB dropped)
    __builtin_prefetch(pb[0] + kk + 32, 0, 1);
    __builtin_prefetch(pb[2] + kk + 32, 0, 1);

#pragma unroll
    for (int m = 0; m < 2; ++m)
#pragma unroll
      for (int n = 0; n < 4; ++n)
        acc[m][n] = __builtin_amdgcn_wmma_f32_16x16x32_bf16(
            false, aF[m].v, false, bF[n].v, (short)0, acc[m][n], false, false);
  }

  // C/D 16x16 f32 layout: VGPR r, lane L -> row r + 8*(L>>4), col L&15
  const int cRowBase = blockM + wm * 32;
  const int cColBase = blockN + wn * 64;
#pragma unroll
  for (int m = 0; m < 2; ++m) {
#pragma unroll
    for (int n = 0; n < 4; ++n) {
#pragma unroll
      for (int r = 0; r < 8; ++r) {
        const int gM = cRowBase + m * 16 + r + 8 * halfSel;
        const int gN = cColBase + n * 16 + l15;
        if (gN < N) {
          float v = acc[m][n][r];
          if (Cf) Cf[(size_t)gM * ldc + gN] = v + bias[gN];
          else    Cb[(size_t)gM * ldc + gN] = f32_to_bf16_bits(v);
        }
      }
    }
  }
}

// ---------------------------------------------------------------------------
// Per-token reduction: one 256-thread block per token.
// Computes cluster logits (h0 . kernel_cluster + bias_cluster), then a
// single-pass online (max, sum-exp) over each logit segment in d_out.
// stats[t] = {M0, Z0(joint incl. cluster), c1, c2, M1, S1, M2, S2}
// ---------------------------------------------------------------------------
static __device__ void online_segment(const float* __restrict__ base, int n,
                                      int tid, float* sm, float* ss,
                                      float& M, float& S) {
  float m = -INFINITY, s = 0.f;
  for (int j = tid; j < n; j += 256) {
    float v = base[j];
    if (v > m) { s = s * __expf(m - v) + 1.f; m = v; }
    else       { s += __expf(v - m); }
  }
  sm[tid] = m; ss[tid] = s;
  __syncthreads();
  for (int off = 128; off > 0; off >>= 1) {
    if (tid < off) {
      float m1 = sm[tid], s1 = ss[tid];
      float m2 = sm[tid + off], s2 = ss[tid + off];
      float mm = fmaxf(m1, m2);
      sm[tid] = mm;
      ss[tid] = s1 * __expf(m1 - mm) + s2 * __expf(m2 - mm);
    }
    __syncthreads();
  }
  M = sm[0]; S = ss[0];
  __syncthreads();
}

__global__ __launch_bounds__(256) void softmax_reduce(
    const float* __restrict__ logits,        // d_out: TOKENS x VT raw logits
    const uint16_t* __restrict__ h0b,        // TOKENS x 512 bf16
    const float* __restrict__ kernel_cluster,// 512 x 2
    const float* __restrict__ bias_cluster,  // 2
    float* __restrict__ stats)               // TOKENS x 8
{
  const int t   = blockIdx.x;
  const int tid = threadIdx.x;
  __shared__ float sm[256];
  __shared__ float ss[256];

  // cluster logits: h0[t] (bf16) . kernel_cluster columns, f32 accumulate
  float p1 = 0.f, p2 = 0.f;
  for (int i = tid; i < IN_DIM; i += 256) {
    float h = bf16_bits_to_f32(h0b[(size_t)t * IN_DIM + i]);
    p1 += h * kernel_cluster[i * 2 + 0];
    p2 += h * kernel_cluster[i * 2 + 1];
  }
  sm[tid] = p1; ss[tid] = p2;
  __syncthreads();
  for (int off = 128; off > 0; off >>= 1) {
    if (tid < off) { sm[tid] += sm[tid + off]; ss[tid] += ss[tid + off]; }
    __syncthreads();
  }
  const float cl1 = sm[0] + bias_cluster[0];
  const float cl2 = ss[0] + bias_cluster[1];
  __syncthreads();

  const float* row = logits + (size_t)t * VT;
  float M0, S0, M1, S1, M2, S2;
  online_segment(row, V0, tid, sm, ss, M0, S0);
  // fold the two cluster logits into the joint softmax stats
  const float mm = fmaxf(M0, fmaxf(cl1, cl2));
  const float Z0 = S0 * __expf(M0 - mm) + __expf(cl1 - mm) + __expf(cl2 - mm);
  online_segment(row + V0,      V1, tid, sm, ss, M1, S1);
  online_segment(row + V0 + V1, V2, tid, sm, ss, M2, S2);

  if (tid == 0) {
    float* st = stats + (size_t)t * 8;
    st[0] = mm; st[1] = Z0;
    st[2] = __expf(cl1 - mm) / Z0;   // cluster prob 1
    st[3] = __expf(cl2 - mm) / Z0;   // cluster prob 2
    st[4] = M1; st[5] = S1;
    st[6] = M2; st[7] = S2;
  }
}

// ---------------------------------------------------------------------------
// In-place normalize: head -> joint softmax; tails -> softmax * cluster prob
// ---------------------------------------------------------------------------
__global__ __launch_bounds__(256) void softmax_normalize(
    float* __restrict__ out, const float* __restrict__ stats)
{
  const int t = blockIdx.y;
  const int j = blockIdx.x * blockDim.x + threadIdx.x;
  if (j >= VT) return;
  const float* st = stats + (size_t)t * 8;
  float* p = out + (size_t)t * VT + j;
  const float v = *p;
  float r;
  if (j < V0)            r = __expf(v - st[0]) / st[1];
  else if (j < V0 + V1)  r = __expf(v - st[4]) / st[5] * st[2];
  else                   r = __expf(v - st[6]) / st[7] * st[3];
  *p = r;
}

// ---------------------------------------------------------------------------
// launcher
// ---------------------------------------------------------------------------
extern "C" void kernel_launch(void* const* d_in, const int* in_sizes, int n_in,
                              void* d_out, int out_size, void* d_ws, size_t ws_size,
                              hipStream_t stream) {
  (void)in_sizes; (void)n_in; (void)out_size; (void)ws_size;

  // dict order: x, emb0, proj0, bias0, emb1, proj1, bias1, emb2, proj2, bias2,
  //             kernel_cluster, bias_cluster
  const float* x     = (const float*)d_in[0];
  const float* emb0  = (const float*)d_in[1];
  const float* proj0 = (const float*)d_in[2];
  const float* bias0 = (const float*)d_in[3];
  const float* emb1  = (const float*)d_in[4];
  const float* proj1 = (const float*)d_in[5];
  const float* bias1 = (const float*)d_in[6];
  const float* emb2  = (const float*)d_in[7];
  const float* proj2 = (const float*)d_in[8];
  const float* bias2 = (const float*)d_in[9];
  const float* kc    = (const float*)d_in[10];
  const float* bc    = (const float*)d_in[11];
  float* out = (float*)d_out;

  // workspace carve-up (all offsets 64B-aligned)
  char* ws = (char*)d_ws;
  size_t off = 0;
  auto take = [&](size_t bytes) { char* p = ws + off; off += bytes; return p; };
  uint16_t* x_b  = (uint16_t*)take((size_t)TOKENS * IN_DIM * 2);
  uint16_t* p0_b = (uint16_t*)take((size_t)512 * 512 * 2);
  uint16_t* p1_b = (uint16_t*)take((size_t)128 * 512 * 2);
  uint16_t* p2_b = (uint16_t*)take((size_t)32  * 512 * 2);
  uint16_t* e0_b = (uint16_t*)take((size_t)V0 * 512 * 2);
  uint16_t* e1_b = (uint16_t*)take((size_t)V1 * 128 * 2);
  uint16_t* e2_b = (uint16_t*)take((size_t)V2 * 32  * 2);
  uint16_t* h0_b = (uint16_t*)take((size_t)TOKENS * 512 * 2);
  uint16_t* h1_b = (uint16_t*)take((size_t)TOKENS * 128 * 2);
  uint16_t* h2_b = (uint16_t*)take((size_t)TOKENS * 32  * 2);
  float*    stats = (float*)take((size_t)TOKENS * 8 * 4);

  auto cvt = [&](const float* s, uint16_t* d, int n) {
    cvt_f32_to_bf16<<<(n + 255) / 256, 256, 0, stream>>>(s, d, n);
  };
  cvt(x,     x_b,  TOKENS * IN_DIM);
  cvt(proj0, p0_b, 512 * 512);
  cvt(proj1, p1_b, 128 * 512);
  cvt(proj2, p2_b, 32  * 512);
  cvt(emb0,  e0_b, V0 * 512);
  cvt(emb1,  e1_b, V1 * 128);
  cvt(emb2,  e2_b, V2 * 32);

  const dim3 blk(256);
  auto gemm = [&](const uint16_t* A, const uint16_t* B, int N, int K,
                  float* Cf, uint16_t* Cb, const float* bias, int ldc) {
    dim3 grid((N + 255) / 256, TOKENS / 64);
    gemm_nt_bf16<<<grid, blk, 0, stream>>>(A, B, N, K, Cf, Cb, bias, ldc);
  };

  // h_i = x @ proj_i^T  (bf16 epilogue into workspace)
  gemm(x_b, p0_b, 512, IN_DIM, nullptr, h0_b, nullptr, 512);
  gemm(x_b, p1_b, 128, IN_DIM, nullptr, h1_b, nullptr, 128);
  gemm(x_b, p2_b, 32,  IN_DIM, nullptr, h2_b, nullptr, 32);

  // logits_i = h_i @ emb_i^T + bias_i  (raw f32 logits straight into d_out)
  gemm(h0_b, e0_b, V0, 512, out,             nullptr, bias0, VT);
  gemm(h1_b, e1_b, V1, 128, out + V0,        nullptr, bias1, VT);
  gemm(h2_b, e2_b, V2, 32,  out + V0 + V1,   nullptr, bias2, VT);

  // per-token softmax statistics (+ cluster logits), then normalize in place
  softmax_reduce<<<TOKENS, 256, 0, stream>>>(out, h0_b, kc, bc, stats);
  softmax_normalize<<<dim3((VT + 255) / 256, TOKENS), 256, 0, stream>>>(out, stats);
}
